// MorphModel_57200374448162
// MI455X (gfx1250) — matrix operand order repaired and checked by reading, not compiled, not run on hardware
//
#include <hip/hip_runtime.h>
#include <hip/hip_bf16.h>
#include <math.h>

typedef __bf16 bf16;
typedef __attribute__((ext_vector_type(16))) __bf16 v16bf;
typedef __attribute__((ext_vector_type(8)))  __bf16 v8bf;
typedef __attribute__((ext_vector_type(8)))  float  v8f;

#define B_TOK   32768
#define D_IN_   256
#define D_HID_  512
#define D_OUT_  256
#define N_EXP_  16
#define TOPK_   2
#define TM_     64                                  // slots per GEMM tile
#define SLOT_CAP   (B_TOK*TOPK_ + N_EXP_*TM_)       // 66560 (padded capacity)
#define N_TILES    (B_TOK*TOPK_/TM_ + N_EXP_)       // 1040 fixed grid

// ---------------------------------------------------------------- gating ----
// one wave (32 lanes) per token; lane owns 8 input features
__global__ __launch_bounds__(256) void k_gate(
    const float* __restrict__ x, const float* __restrict__ Wg,
    const float* __restrict__ bg, bf16* __restrict__ xb,
    int* __restrict__ pairE, float* __restrict__ pairW, int* __restrict__ counts)
{
    int lane = threadIdx.x & 31;
    int wv   = threadIdx.x >> 5;
    int b    = blockIdx.x * 8 + wv;

    const float* xr = x + (size_t)b * D_IN_ + lane * 8;
    float xv[8];
#pragma unroll
    for (int j = 0; j < 8; j++) xv[j] = xr[j];

    // write bf16 copy of x for the WMMA phase
    bf16* xo = xb + (size_t)b * D_IN_ + lane * 8;
#pragma unroll
    for (int j = 0; j < 8; j++) xo[j] = (bf16)xv[j];

    float lg[N_EXP_];
#pragma unroll
    for (int e = 0; e < N_EXP_; e++) lg[e] = 0.f;
#pragma unroll
    for (int j = 0; j < 8; j++) {
        const float* wr = Wg + (size_t)(lane * 8 + j) * N_EXP_;
#pragma unroll
        for (int e = 0; e < N_EXP_; e++) lg[e] += xv[j] * wr[e];
    }
    // wave32 butterfly reduce -> every lane has full logits
#pragma unroll
    for (int off = 16; off >= 1; off >>= 1) {
#pragma unroll
        for (int e = 0; e < N_EXP_; e++) lg[e] += __shfl_xor(lg[e], off, 32);
    }

    float m1 = -1e30f, m2 = -1e30f; int i1 = 0, i2 = 0;
#pragma unroll
    for (int e = 0; e < N_EXP_; e++) {
        float v = lg[e] + bg[e];
        if (v > m1)      { m2 = m1; i2 = i1; m1 = v; i1 = e; }
        else if (v > m2) { m2 = v;  i2 = e; }
    }
    // renormalized top-2 softmax: denominator over all experts cancels
    float w2 = __expf(m2 - m1);
    float inv = 1.f / (1.f + w2);
    if (lane == 0) {
        pairE[2 * b + 0] = i1; pairE[2 * b + 1] = i2;
        pairW[2 * b + 0] = inv; pairW[2 * b + 1] = w2 * inv;
        atomicAdd(&counts[i1], 1);
        atomicAdd(&counts[i2], 1);
    }
}

// --------------------------------------------------------------- helpers ----
__global__ void k_init_counts(int* counts) {
    if (threadIdx.x < N_EXP_) counts[threadIdx.x] = 0;
}

__global__ void k_scan(const int* __restrict__ counts, int* __restrict__ offs,
                       int* __restrict__ cursors) {
    if (threadIdx.x == 0) {
        int o = 0;
        for (int e = 0; e < N_EXP_; e++) {
            offs[e] = o; cursors[e] = o;
            o += (counts[e] + TM_ - 1) & ~(TM_ - 1);   // pad each expert to tile
        }
        offs[N_EXP_] = o;
    }
}

__global__ __launch_bounds__(256) void k_fill_slots(int* __restrict__ token_ids,
                                                    float* __restrict__ token_w) {
    int i = blockIdx.x * 256 + threadIdx.x;
    if (i < SLOT_CAP) { token_ids[i] = -1; token_w[i] = 0.f; }
}

__global__ __launch_bounds__(256) void k_scatter(
    const int* __restrict__ pairE, const float* __restrict__ pairW,
    int* __restrict__ cursors, int* __restrict__ token_ids, float* __restrict__ token_w)
{
    int b = blockIdx.x * 256 + threadIdx.x;
    if (b >= B_TOK) return;
#pragma unroll
    for (int k = 0; k < TOPK_; k++) {
        int e = pairE[2 * b + k];
        int pos = atomicAdd(&cursors[e], 1);
        token_ids[pos] = b;
        token_w[pos]  = pairW[2 * b + k];
    }
}

// W1[e][d][h] -> W1T[e][h][d] bf16 (B-fragment wants contiguous K per column)
__global__ __launch_bounds__(256) void k_cvt_w1(const float* __restrict__ W1,
                                                bf16* __restrict__ W1T) {
    int i = blockIdx.x * 256 + threadIdx.x;          // 16*256*512 = 2M exact
    int e = i >> 17, rem = i & 131071;
    int d = rem >> 9, h = rem & 511;
    W1T[((size_t)e * D_HID_ + h) * D_IN_ + d] = (bf16)W1[i];
}
// W2[e][h][o] -> W2T[e][o][h] bf16
__global__ __launch_bounds__(256) void k_cvt_w2(const float* __restrict__ W2,
                                                bf16* __restrict__ W2T) {
    int i = blockIdx.x * 256 + threadIdx.x;
    int e = i >> 17, rem = i & 131071;
    int h = rem >> 8, o = rem & 255;
    W2T[((size_t)e * D_OUT_ + o) * D_HID_ + h] = (bf16)W2[i];
}

__global__ __launch_bounds__(256) void k_zero_out(float4* __restrict__ out) {
    out[blockIdx.x * 256 + threadIdx.x] = float4{0.f, 0.f, 0.f, 0.f};
}

// load 16 bf16 forming one A-fragment lane slice: [k0+hi*8 .. +7] and [k0+16+hi*8 .. +7]
__device__ __forceinline__ v16bf ldfrag(const bf16* p0, const bf16* p1) {
    v8bf lo = *reinterpret_cast<const v8bf*>(p0);
    v8bf hi = *reinterpret_cast<const v8bf*>(p1);
    v16bf a;
#pragma unroll
    for (int i = 0; i < 8; i++) { a[i] = lo[i]; a[i + 8] = hi[i]; }
    return a;
}

// --------------------------------------------------- fused expert MLP -------
// grid: N_TILES blocks x 512 threads (16 waves). Each block: 64 slots of one expert.
// ks is the OUTER compute loop: A-frags loaded once per ks and consumed against
// all of the wave's B-tiles immediately -> register reuse by construction,
// accumulators persist in VGPRs, no >256-VGPR blowup / s_set_vgpr_msb spam.
__global__ __launch_bounds__(512) void k_moe(
    const bf16* __restrict__ xb, const bf16* __restrict__ W1T, const bf16* __restrict__ W2T,
    const float* __restrict__ b1, const float* __restrict__ b2,
    const int* __restrict__ offs, const int* __restrict__ token_ids,
    const float* __restrict__ token_w, float* __restrict__ out)
{
    __shared__ __align__(16) bf16 Xs[TM_][D_IN_ + 8];    // +8 elems: bank spread
    __shared__ __align__(16) bf16 Hs[TM_][D_HID_ + 8];

    // map flat tile id -> (expert, slot base)
    int tile = blockIdx.x;
    int e = -1, base = 0, cum = 0;
#pragma unroll
    for (int i = 0; i < N_EXP_; i++) {
        int t = (offs[i + 1] - offs[i]) >> 6;
        if (e < 0 && tile < cum + t) { e = i; base = offs[i] + (tile - cum) * TM_; }
        cum += t;
    }
    if (e < 0) return;                                   // beyond last used tile

    const int tid  = threadIdx.x;
    const int lane = tid & 31, wave = tid >> 5;          // 16 waves
    const int l = lane & 15, hi = lane >> 4;
    const int hi8 = hi * 8, hi16 = hi * 16;
    const int mhalf = wave & 1;                          // M rows [mhalf*32, +32)
    const int wquad = wave >> 1;                         // 0..7

    // gather X tile (64 rows x 256 bf16) into LDS, zero for padded slots
    for (int i = tid; i < TM_ * (D_IN_ / 8); i += 512) {
        int row = i >> 5, seg = i & 31;
        int tok = token_ids[base + row];
        uint4 v = uint4{0u, 0u, 0u, 0u};
        if (tok >= 0)
            v = *reinterpret_cast<const uint4*>(xb + (size_t)tok * D_IN_ + seg * 8);
        *reinterpret_cast<uint4*>(&Xs[row][seg * 8]) = v;
    }
    __syncthreads();

    // ---------------- GEMM1: H = relu(X @ W1[e] + b1[e]) -------------------
    // wave tile: rows [mhalf*32, +32) x hidden cols [wquad*64, +64)
    {
        const int arow0 = mhalf * 32 + l;
        const int ncol0 = wquad * 64 + l;
        v8f z = {};
        v8f acc[8];                                      // [nt][mt], 64 VGPRs
#pragma unroll
        for (int i = 0; i < 8; i++) acc[i] = z;

        const bf16* wbase = W1T + ((size_t)e * D_HID_ + ncol0) * D_IN_;
#pragma unroll 2
        for (int ks = 0; ks < D_IN_ / 32; ks++) {
            const int k0 = ks * 32;
            v16bf a0 = ldfrag(&Xs[arow0][k0 + hi8],      &Xs[arow0][k0 + 16 + hi8]);
            v16bf a1 = ldfrag(&Xs[arow0 + 16][k0 + hi8], &Xs[arow0 + 16][k0 + 16 + hi8]);
#pragma unroll
            for (int nt = 0; nt < 4; nt++) {
                v16bf bfrag = ldfrag(wbase + (size_t)nt * 16 * D_IN_ + k0 + hi8,
                                     wbase + (size_t)nt * 16 * D_IN_ + k0 + 16 + hi8);
                acc[nt * 2 + 0] = __builtin_amdgcn_wmma_f32_16x16x32_bf16(
                    false, a0, false, bfrag, (short)0, acc[nt * 2 + 0], false, false);
                acc[nt * 2 + 1] = __builtin_amdgcn_wmma_f32_16x16x32_bf16(
                    false, a1, false, bfrag, (short)0, acc[nt * 2 + 1], false, false);
            }
        }
#pragma unroll
        for (int nt = 0; nt < 4; nt++) {
            const int ncol = wquad * 64 + nt * 16 + l;
            const float bias = b1[e * D_HID_ + ncol];
#pragma unroll
            for (int mt = 0; mt < 2; mt++)
#pragma unroll
                for (int r = 0; r < 8; r++)
                    Hs[mhalf * 32 + mt * 16 + hi8 + r][ncol] =
                        (bf16)fmaxf(acc[nt * 2 + mt][r] + bias, 0.f);
        }
    }
    __syncthreads();

    // ---------------- GEMM2: Y = H @ W2[e] + b2[e], weighted scatter -------
    // wave tile: rows [mhalf*32, +32) x out cols [wquad*32, +32)
    {
        const int arow0 = mhalf * 32 + l;
        const int ncol0 = wquad * 32 + l;
        v8f z = {};
        v8f acc[4];                                      // [nt][mt], 32 VGPRs
#pragma unroll
        for (int i = 0; i < 4; i++) acc[i] = z;

        const bf16* wbase = W2T + ((size_t)e * D_OUT_ + ncol0) * D_HID_;
#pragma unroll 2
        for (int ks = 0; ks < D_HID_ / 32; ks++) {
            const int k0 = ks * 32;
            v16bf a0 = ldfrag(&Hs[arow0][k0 + hi8],      &Hs[arow0][k0 + 16 + hi8]);
            v16bf a1 = ldfrag(&Hs[arow0 + 16][k0 + hi8], &Hs[arow0 + 16][k0 + 16 + hi8]);
#pragma unroll
            for (int nt = 0; nt < 2; nt++) {
                v16bf bfrag = ldfrag(wbase + (size_t)nt * 16 * D_HID_ + k0 + hi8,
                                     wbase + (size_t)nt * 16 * D_HID_ + k0 + 16 + hi8);
                acc[nt * 2 + 0] = __builtin_amdgcn_wmma_f32_16x16x32_bf16(
                    false, a0, false, bfrag, (short)0, acc[nt * 2 + 0], false, false);
                acc[nt * 2 + 1] = __builtin_amdgcn_wmma_f32_16x16x32_bf16(
                    false, a1, false, bfrag, (short)0, acc[nt * 2 + 1], false, false);
            }
        }
#pragma unroll
        for (int nt = 0; nt < 2; nt++) {
            const int ncol = wquad * 32 + nt * 16 + l;
            const float bias = b2[e * D_OUT_ + ncol];
#pragma unroll
            for (int mt = 0; mt < 2; mt++) {
#pragma unroll
                for (int r = 0; r < 8; r++) {
                    int slot = base + mhalf * 32 + mt * 16 + hi8 + r;
                    int tok  = token_ids[slot];
                    if (tok >= 0) {
                        float w = token_w[slot];
                        atomicAdd(&out[(size_t)tok * D_OUT_ + ncol],
                                  w * (acc[nt * 2 + mt][r] + bias));
                    }
                }
            }
        }
    }
}

// ---------------------------------------------------------------- launch ----
extern "C" void kernel_launch(void* const* d_in, const int* in_sizes, int n_in,
                              void* d_out, int out_size, void* d_ws, size_t ws_size,
                              hipStream_t stream) {
    const float* x  = (const float*)d_in[0];
    const float* Wg = (const float*)d_in[1];
    const float* bg = (const float*)d_in[2];
    const float* W1 = (const float*)d_in[3];
    const float* b1 = (const float*)d_in[4];
    const float* W2 = (const float*)d_in[5];
    const float* b2 = (const float*)d_in[6];
    float* out = (float*)d_out;

    char* ws = (char*)d_ws;
    size_t o = 0;
    bf16*  xb   = (bf16*)(ws + o);  o += (size_t)B_TOK * D_IN_ * 2;           // 16 MB
    bf16*  W1T  = (bf16*)(ws + o);  o += (size_t)N_EXP_ * D_IN_ * D_HID_ * 2; // 4 MB
    bf16*  W2T  = (bf16*)(ws + o);  o += (size_t)N_EXP_ * D_HID_ * D_OUT_ * 2;// 4 MB
    int*   meta = (int*)(ws + o);   o += 256;                                 // counts/offs/cursors
    int*   counts  = meta;
    int*   offs    = meta + 16;
    int*   cursors = meta + 33;
    int*   pairE = (int*)(ws + o);   o += (size_t)B_TOK * 2 * 4;
    float* pairW = (float*)(ws + o); o += (size_t)B_TOK * 2 * 4;
    int*   token_ids = (int*)(ws + o);   o += (size_t)SLOT_CAP * 4;
    float* token_w   = (float*)(ws + o); o += (size_t)SLOT_CAP * 4;

    k_init_counts<<<1, 32, 0, stream>>>(counts);
    k_cvt_w1<<<8192, 256, 0, stream>>>(W1, W1T);
    k_cvt_w2<<<8192, 256, 0, stream>>>(W2, W2T);
    k_gate<<<B_TOK / 8, 256, 0, stream>>>(x, Wg, bg, xb, pairE, pairW, counts);
    k_scan<<<1, 1, 0, stream>>>(counts, offs, cursors);
    k_fill_slots<<<(SLOT_CAP + 255) / 256, 256, 0, stream>>>(token_ids, token_w);
    k_scatter<<<B_TOK / 256, 256, 0, stream>>>(pairE, pairW, cursors, token_ids, token_w);
    k_zero_out<<<(B_TOK * D_OUT_) / 1024, 256, 0, stream>>>((float4*)out);
    k_moe<<<N_TILES, 512, 0, stream>>>(xb, W1T, W2T, b1, b2, offs,
                                       token_ids, token_w, out);
}
